// FaFMIMONet_512_16_16_KD_31963146617588
// MI455X (gfx1250) — matrix operand order repaired and checked by reading, not compiled, not run on hardware
//
#include <hip/hip_runtime.h>
#include <hip/hip_bf16.h>

// FaF MIMO fusion for MI455X (gfx1250, wave32).
//
// Each pair-warp (two chained bilinear grid_samples) is a channel-independent
// linear operator on the 256 spatial positions -> expressed as a dense GEMM
//   out[64ch x 256px] = F_j[64ch x 256q] * M[256q x 256px]
// where M (<=16 nnz/column) is built in LDS per (b,i,j) and the GEMM runs on
// v_wmma_f32_16x16x32_f16. Memory-bound target: ~84MB / 23.3TB/s ~= 3.6us.

typedef __attribute__((ext_vector_type(16))) _Float16 v16h;
typedef __attribute__((ext_vector_type(8)))  _Float16 v8h;
typedef __attribute__((ext_vector_type(8)))  float    v8f;

#define BB 16
#define AA 5
#define CC 512
#define PP 256            // H*W = 16*16
#define ROWPAD 264        // padded LDS row length in halfs (breaks 512B bank stride)
#define MCHUNK 64         // channels per workgroup
#define NMC (CC / MCHUNK) // 8 channel chunks

#define MT_HALFS (PP * ROWPAD)      // composed warp matrix Mt[p'][q]
#define AF_HALFS (MCHUNK * ROWPAD)  // staged f16 feature block Af[ch][q]
#define SMEM_BYTES (size_t)((MT_HALFS + AF_HALFS) * 2)

__global__ void fafmimo_fuse_wmma(const float* __restrict__ feat,
                                  const float* __restrict__ trans,
                                  const int*   __restrict__ numag,
                                  float*       __restrict__ out) {
  extern __shared__ _Float16 smem[];
  _Float16* Mt = smem;            // [256][ROWPAD]  B-operand (q-major per row p')
  _Float16* Af = smem + MT_HALFS; // [64][ROWPAD]   A-operand (q-major per channel)

  const int blk = blockIdx.x;
  const int mc  = blk % NMC;
  const int ai  = (blk / NMC) % AA;
  const int b   = blk / (NMC * AA);
  const int c0  = mc * MCHUNK;

  const int tid  = threadIdx.x;     // 0..255, also the output pixel p' this thread owns
  const int lane = tid & 31;
  const int wave = tid >> 5;        // 8 waves
  const int mt   = wave >> 1;       // m-tile 0..3 (16 channels each)
  const int nh   = wave & 1;        // n-half: tiles [0..7] or [8..15]
  const int nlo  = lane & 15;
  const int lhi  = lane >> 4;

  const int n = numag[b * AA];      // num_agent_tensor[b][0]

  v8f acc[8];
  v8f zero = {};
#pragma unroll
  for (int t = 0; t < 8; ++t) acc[t] = zero;

  if (ai < n) {
    for (int j = 0; j < AA; ++j) {
      if (j == ai || j >= n) continue;

      const float* fj = feat + (((size_t)j * BB + b) * CC + c0) * PP;
      // Pull the 64KB neighbor block toward the caches (global_prefetch_b8).
      __builtin_prefetch(fj + tid * 64, 0, 1);

      // ---- 1. zero this thread's private row of Mt (528B = 33 x 16B) ----
      {
        uint4* row = (uint4*)&Mt[tid * ROWPAD];
        uint4 z4 = make_uint4(0u, 0u, 0u, 0u);
#pragma unroll
        for (int k = 0; k < 33; ++k) row[k] = z4;
      }

      // ---- 2. build column p'=tid of the composed warp matrix ----
      // translation sample: src = (px + T03/4, py - T13/4); rotation sample at
      // each translation tap location.  Mt[p'][q] += tw * rw for each of the
      // <=16 in-bounds composed taps (thread-private row: no races).
      {
        const float* T = trans + (((size_t)b * AA + ai) * AA + j) * 16;
        const float r00 = T[0], r01 = T[1], t03 = T[3];
        const float r10 = T[4], r11 = T[5], t13 = T[7];
        const int px = tid & 15, py = tid >> 4;
        const float sx = (float)px + 0.25f * t03;
        const float sy = (float)py - 0.25f * t13;
        const float fx0 = floorf(sx), fy0 = floorf(sy);
        const float wx1 = sx - fx0, wx0 = 1.0f - wx1;
        const float wy1 = sy - fy0, wy0 = 1.0f - wy1;
#pragma unroll
        for (int tk = 0; tk < 4; ++tk) {
          const float xi = fx0 + (float)(tk & 1);
          const float yi = fy0 + (float)(tk >> 1);
          const float tw = ((tk & 1) ? wx1 : wx0) * ((tk >> 1) ? wy1 : wy0);
          if (xi < 0.0f || xi > 15.0f || yi < 0.0f || yi > 15.0f) continue;
          const float xs = (2.0f * xi + 1.0f) * 0.0625f - 1.0f;
          const float ys = (2.0f * yi + 1.0f) * 0.0625f - 1.0f;
          const float ix = 8.0f * (r00 * xs + r01 * ys) + 7.5f;
          const float iy = 8.0f * (r10 * xs + r11 * ys) + 7.5f;
          const float gx0 = floorf(ix), gy0 = floorf(iy);
          const float rx1 = ix - gx0, rx0 = 1.0f - rx1;
          const float ry1 = iy - gy0, ry0 = 1.0f - ry1;
#pragma unroll
          for (int rm = 0; rm < 4; ++rm) {
            const float qx = gx0 + (float)(rm & 1);
            const float qy = gy0 + (float)(rm >> 1);
            if (qx < 0.0f || qx > 15.0f || qy < 0.0f || qy > 15.0f) continue;
            const float w = tw * ((rm & 1) ? rx1 : rx0) * ((rm >> 1) ? ry1 : ry0);
            const int q = (int)qy * 16 + (int)qx;
            const int idx = tid * ROWPAD + q;
            Mt[idx] = (_Float16)((float)Mt[idx] + w);
          }
        }
      }

      // ---- 3. stage the 64ch x 256px neighbor feature block, f32 -> f16 ----
      for (int v = tid; v < (MCHUNK * PP) / 4; v += 256) {
        const int le  = v * 4;
        const int chl = le >> 8;
        const int q   = le & 255;
        const float4 d = *(const float4*)(fj + le);
        _Float16* dst = &Af[chl * ROWPAD + q];
        dst[0] = (_Float16)d.x; dst[1] = (_Float16)d.y;
        dst[2] = (_Float16)d.z; dst[3] = (_Float16)d.w;
      }
      __syncthreads();

      // ---- 4. GEMM: acc[64x256] += Af[64x256] * Mt'[256x256] on WMMA ----
      // A fragment (16x32 f16): lanes 0-15 row=nlo, K {k..k+7, k+16..k+23};
      //                         lanes 16-31 row=nlo, K {k+8..k+15, k+24..k+31}.
      // B fragment (32x16 f16): lane column = nlo, K order same as A fragment.
      const int arow = mt * 16 + nlo;
      for (int kt = 0; kt < 8; ++kt) {
        const int aq = kt * 32 + (lhi ? 8 : 0);
        const v8h alo = *(const v8h*)&Af[arow * ROWPAD + aq];
        const v8h ahi = *(const v8h*)&Af[arow * ROWPAD + aq + 16];
        v16h a;
#pragma unroll
        for (int e = 0; e < 8; ++e) { a[e] = alo[e]; a[e + 8] = ahi[e]; }
        const int bq = kt * 32 + (lhi ? 8 : 0);
#pragma unroll
        for (int t = 0; t < 8; ++t) {
          const int prow = (nh * 8 + t) * 16 + nlo;
          const v8h blo = *(const v8h*)&Mt[prow * ROWPAD + bq];
          const v8h bhi = *(const v8h*)&Mt[prow * ROWPAD + bq + 16];
          v16h bm;
#pragma unroll
          for (int e = 0; e < 8; ++e) { bm[e] = blo[e]; bm[e + 8] = bhi[e]; }
          acc[t] = __builtin_amdgcn_wmma_f32_16x16x32_f16(
              false, a, false, bm, (short)0, acc[t], false, false);
        }
      }
      __syncthreads();  // Mt/Af reused next j
    }
  }

  // ---- 5. epilogue: out = local_i + acc (full coverage of d_out) ----
  // C/D layout: VGPR r -> M = r (lanes 0-15) / 8+r (lanes 16-31); N = lane%16.
  const size_t obase = (((size_t)ai * BB + b) * CC + c0) * PP;
#pragma unroll
  for (int t = 0; t < 8; ++t) {
    const int p = (nh * 8 + t) * 16 + nlo;
#pragma unroll
    for (int r = 0; r < 8; ++r) {
      const int ch = mt * 16 + lhi * 8 + r;
      const size_t off = obase + (size_t)ch * PP + p;
      out[off] = feat[off] + acc[t][r];
    }
  }
}

extern "C" void kernel_launch(void* const* d_in, const int* in_sizes, int n_in,
                              void* d_out, int out_size, void* d_ws, size_t ws_size,
                              hipStream_t stream) {
  const float* feat  = (const float*)d_in[0];  // [A*B, C, 16, 16] f32
  const float* trans = (const float*)d_in[1];  // [B, A, A, 4, 4]  f32
  const int*   numag = (const int*)d_in[2];    // [B, A]           i32
  float* out = (float*)d_out;                  // [A*B, C, 16, 16] f32
  (void)in_sizes; (void)n_in; (void)out_size; (void)d_ws; (void)ws_size;

  dim3 grid(BB * AA * NMC);  // 640 workgroups: (b, agent_i, 64-ch chunk)
  dim3 block(256);           // 8 wave32s
  fafmimo_fuse_wmma<<<grid, block, SMEM_BYTES, stream>>>(feat, trans, numag, out);
}
